// MEGNet_47974784696351
// MI455X (gfx1250) — matrix operand order repaired and checked by reading, not compiled, not run on hardware
//
#include <hip/hip_runtime.h>
#include <stdint.h>
#include <stddef.h>

typedef __bf16 bf16;
typedef __attribute__((ext_vector_type(16))) __bf16 v16bf;
typedef __attribute__((ext_vector_type(8)))  float  v8f;
typedef unsigned int u32x4 __attribute__((ext_vector_type(4)));

#define NNODES 50000
#define NEDGES 500000
#define NGRAPH 256

static __device__ __forceinline__ bf16 f2b(float f) {
  union { float f; uint32_t u; } a; a.f = f;
  uint32_t u = a.u;
  uint32_t r = (u + 0x7fffu + ((u >> 16) & 1u)) >> 16;   // RNE f32 -> bf16
  union { uint16_t u; bf16 h; } b; b.u = (uint16_t)r;
  return b.h;
}

// convert f32 (rows x kin) -> bf16 (rows x 64), zero-padded columns kin..63
__global__ void k_f2b_pad(const float* __restrict__ x, bf16* __restrict__ y,
                          long rows, int kin) {
  long i = (long)blockIdx.x * blockDim.x + threadIdx.x;
  if (i >= rows * 64) return;
  long r = i >> 6; int c = (int)(i & 63);
  float v = (c < kin) ? x[r * kin + c] : 0.f;
  y[i] = f2b(v);
}

__global__ void k_zero(float* __restrict__ p, long n) {
  long i = (long)blockIdx.x * blockDim.x + threadIdx.x;
  if (i < n) p[i] = 0.f;
}

__global__ void k_split_edges(const int* __restrict__ ei, const int* __restrict__ gidx,
                              int* __restrict__ src, int* __restrict__ dst,
                              int* __restrict__ egi, int E) {
  int e = blockIdx.x * blockDim.x + threadIdx.x;
  if (e >= E) return;
  int s = ei[2 * e], d = ei[2 * e + 1];
  src[e] = s; dst[e] = d; egi[e] = gidx[s];
}

// Pack fp32 W (K x 64, row-major) into WMMA-B lane layout, K zero-padded to Kp.
// i = ((q*4 + t)*32 + lane)*16 + j ; element = W[q*32 + 8*(lane>>4) + (j&7) + 16*(j>>3)][t*16 + (lane&15)]
__global__ void k_pack_w(const float* __restrict__ W, bf16* __restrict__ dst, int K, int Kp) {
  int i = blockIdx.x * blockDim.x + threadIdx.x;
  int total = Kp * 64;
  if (i >= total) return;
  int j = i & 15, lane = (i >> 4) & 31, t = (i >> 9) & 3, q = i >> 11;
  int h = lane >> 4, n = t * 16 + (lane & 15);
  int k = q * 32 + h * 8 + (j & 7) + 16 * (j >> 3);
  dst[i] = f2b((k < K) ? W[k * 64 + n] : 0.f);
}

__global__ void k_scat(const float* __restrict__ x, const int* __restrict__ ids,
                       float* __restrict__ sums, float* __restrict__ cnt, int M) {
  long i = (long)blockIdx.x * blockDim.x + threadIdx.x;
  if (i >= (long)M * 64) return;
  int r = (int)(i >> 6), f = (int)(i & 63);
  int s = ids[r];
  atomicAdd(&sums[(long)s * 64 + f], x[i]);
  if (f == 0) atomicAdd(&cnt[s], 1.0f);
}

__global__ void k_norm(const float* __restrict__ sums, const float* __restrict__ cnt,
                       float* __restrict__ outF, bf16* __restrict__ outB, int S) {
  long i = (long)blockIdx.x * blockDim.x + threadIdx.x;
  if (i >= (long)S * 64) return;
  int s = (int)(i >> 6);
  float v = sums[i] / fmaxf(cnt[s], 1.0f);
  if (outF) outF[i] = v;
  if (outB) outB[i] = f2b(v);
}

__global__ void k_resid(float* __restrict__ d, const float* __restrict__ a,
                        bf16* __restrict__ db, long n) {
  long i = (long)blockIdx.x * blockDim.x + threadIdx.x;
  if (i >= n) return;
  float v = d[i] + a[i];
  d[i] = v; db[i] = f2b(v);
}

__global__ void k_head3(const float* __restrict__ h, const float* __restrict__ W3,
                        const float* __restrict__ b3, float* __restrict__ out, int G) {
  int g = blockIdx.x * blockDim.x + threadIdx.x;
  if (g >= G) return;
  float s = b3[0];
  for (int k = 0; k < 64; k++) s += h[g * 64 + k] * W3[k];
  out[g] = s;
}

// -------- fused gather/concat GEMM: N = 64, every segment 64 wide, bf16 WMMA ----------
// All call sites use ReLU and M % 16 == 0. OUTMODE: bit0 -> f32 out, bit1 -> bf16 out.
struct Seg { const bf16* base; const int* idx; };
struct GemmP {
  Seg seg[4];
  const bf16* Wp;      // packed weights, (NSEG*64) x 64 in WMMA-B lane layout
  const float* bias;   // 64 (never null)
  float* outF; bf16* outB;
  int M;
};

template <int NSEG, int OUTMODE>
__global__ __launch_bounds__(128) void k_gemm(GemmP p) {
  constexpr int NQ = NSEG * 2;                    // K steps of 32
  constexpr int WELEMS = NSEG * 64 * 64;          // packed weight elements (bf16)
  __shared__ bf16 sW[WELEMS];                     // NSEG * 8KB <= 32KB

  // cooperative global -> LDS copy of packed weights (16B per thread per iter)
  for (int i = threadIdx.x * 8; i < WELEMS; i += 128 * 8)
    *(u32x4*)(sW + i) = *(const u32x4*)(p.Wp + i);
  __syncthreads();

  const int lane = threadIdx.x & 31;
  const int wave = threadIdx.x >> 5;
  int m0 = (blockIdx.x * 4 + wave) * 16;
  const bool active = m0 < p.M;
  if (!active) m0 = 0;                            // keep wave alive, EXEC all-ones
  const int h = lane >> 4, l15 = lane & 15;
  int m = m0 + l15;

  // gathered row base pointers, one per 64-wide concat segment (compile-time count)
  const bf16* rowp[NSEG];
  #pragma unroll
  for (int s = 0; s < NSEG; s++) {
    int r = p.seg[s].idx ? p.seg[s].idx[m] : m;
    rowp[s] = p.seg[s].base + (long)r * 64;
  }

  // ---- double-buffered, fully-unrolled K pipeline ----
  v16bf Abuf[2];
  v16bf Bbuf[2][4];

  auto loadA = [&](int q, v16bf& A) {
    const bf16* rp = rowp[q >> 1];                // constant index after unroll
    const int c0 = (q & 1) * 32 + 8 * h;
    union AU { u32x4 v; bf16 e[8]; } lo, hi;
    lo.v = *(const u32x4*)(rp + c0);
    hi.v = *(const u32x4*)(rp + c0 + 16);
    #pragma unroll
    for (int j = 0; j < 8; j++) { A[j] = lo.e[j]; A[8 + j] = hi.e[j]; }
  };
  auto loadB = [&](int q, v16bf* B) {
    const bf16* wq = sW + (q * 4 * 32 + lane) * 16;
    #pragma unroll
    for (int t = 0; t < 4; t++) B[t] = *(const v16bf*)(wq + t * 32 * 16);
  };

  loadA(0, Abuf[0]);
  loadB(0, Bbuf[0]);

  v8f acc[4] = {};
  #pragma unroll
  for (int q = 0; q < NQ; q++) {
    const int cur = q & 1, nxt = cur ^ 1;
    if (q + 1 < NQ) {                             // prefetch next step while q computes
      loadA(q + 1, Abuf[nxt]);
      loadB(q + 1, Bbuf[nxt]);
    }
    #pragma unroll
    for (int t = 0; t < 4; t++)
      acc[t] = __builtin_amdgcn_wmma_f32_16x16x32_bf16(
                 false, Abuf[cur], false, Bbuf[cur][t], (short)0, acc[t], false, false);
  }

  if (!active) return;                            // wave-uniform; after all WMMAs
  #pragma unroll
  for (int t = 0; t < 4; t++) {
    int n = t * 16 + l15;
    float bn = p.bias[n];
    #pragma unroll
    for (int r = 0; r < 8; r++) {
      int row = m0 + h * 8 + r;                   // M % 16 == 0 -> always in range
      float v = fmaxf(acc[t][r] + bn, 0.f);       // ReLU (all layers)
      long o = (long)row * 64 + n;
      if (OUTMODE & 1) p.outF[o] = v;
      if (OUTMODE & 2) p.outB[o] = f2b(v);
    }
  }
}

// ------------------------------------ host side ---------------------------------------
static inline int cdiv(long a, long b) { return (int)((a + b - 1) / b); }

extern "C" void kernel_launch(void* const* d_in, const int* in_sizes, int n_in,
                              void* d_out, int out_size, void* d_ws, size_t ws_size,
                              hipStream_t stream) {
  (void)in_sizes; (void)out_size; (void)ws_size;
  if (n_in < 59) return;
  auto F = [&](int i) { return (const float*)d_in[i]; };
  const float* node_features = F(0);
  const float* edge_features = F(1);
  const int* edge_indices   = (const int*)d_in[2];
  const int* graph_indices  = (const int*)d_in[3];
  const float* global_features = F(4);
  // params (insertion order): 5..8 emb_node(W1,b1,W2,b2), 9..12 emb_edge, 13..16 emb_glob,
  // 17+12i: blk{i}_edge(4), blk{i}_node(4), blk{i}_glob(4); 53..58 out_W1,b1,W2,b2,W3,b3

  size_t off = 0;
  auto alloc = [&](size_t bytes) -> char* {
    char* p = (char*)d_ws + off;
    off += (bytes + 255) & ~(size_t)255;
    return p;
  };
  // pack W (K x 64) to Kp = 64*NSEG rows (zero-padded)
  auto pack = [&](const float* W, int K, int Kp) -> bf16* {
    bf16* dst = (bf16*)alloc((size_t)Kp * 64 * 2);
    int total = Kp * 64;
    k_pack_w<<<cdiv(total, 256), 256, 0, stream>>>(W, dst, K, Kp);
    return dst;
  };

  // packed weights
  bf16* pw_en1 = pack(F(5), 64, 64);   bf16* pw_en2 = pack(F(7), 64, 64);
  bf16* pw_ee1 = pack(F(9), 32, 64);   bf16* pw_ee2 = pack(F(11), 64, 64);
  bf16* pw_eg1 = pack(F(13), 16, 64);  bf16* pw_eg2 = pack(F(15), 64, 64);
  bf16 *pw_be1[3], *pw_be2[3], *pw_bn1[3], *pw_bn2[3], *pw_bg1[3], *pw_bg2[3];
  for (int i = 0; i < 3; i++) {
    int pb = 17 + 12 * i;
    pw_be1[i] = pack(F(pb + 0), 256, 256); pw_be2[i] = pack(F(pb + 2), 64, 64);
    pw_bn1[i] = pack(F(pb + 4), 192, 192); pw_bn2[i] = pack(F(pb + 6), 64, 64);
    pw_bg1[i] = pack(F(pb + 8), 192, 192); pw_bg2[i] = pack(F(pb + 10), 64, 64);
  }
  bf16* pw_o1 = pack(F(53), 192, 192);  bf16* pw_o2 = pack(F(55), 64, 64);

  // workspace buffers (all activation buffers are 64 columns wide)
  bf16*  nfin_b = (bf16*)alloc((size_t)NNODES * 64 * 2);
  bf16*  efin_b = (bf16*)alloc((size_t)NEDGES * 64 * 2);
  bf16*  gfin_b = (bf16*)alloc((size_t)NGRAPH * 64 * 2);
  float* nf_f   = (float*)alloc((size_t)NNODES * 64 * 4);
  bf16*  nf_b   = (bf16*)alloc((size_t)NNODES * 64 * 2);
  float* ef_f   = (float*)alloc((size_t)NEDGES * 64 * 4);
  bf16*  ef_b   = (bf16*)alloc((size_t)NEDGES * 64 * 2);
  float* gf_f   = (float*)alloc((size_t)NGRAPH * 64 * 4);
  bf16*  gf_b   = (bf16*)alloc((size_t)NGRAPH * 64 * 2);
  bf16*  hid_b  = (bf16*)alloc((size_t)NEDGES * 64 * 2);
  float* e_new  = (float*)alloc((size_t)NEDGES * 64 * 4);
  float* n_new  = (float*)alloc((size_t)NNODES * 64 * 4);
  float* g_new  = (float*)alloc((size_t)NGRAPH * 64 * 4);
  float* nsum   = (float*)alloc((size_t)NNODES * 64 * 4);
  float* ncnt   = (float*)alloc((size_t)NNODES * 4);
  bf16*  msgs_b = (bf16*)alloc((size_t)NNODES * 64 * 2);
  float* gsumA  = (float*)alloc((size_t)NGRAPH * 64 * 4);
  float* gcntA  = (float*)alloc((size_t)NGRAPH * 4);
  float* gsumB  = (float*)alloc((size_t)NGRAPH * 64 * 4);
  float* gcntB  = (float*)alloc((size_t)NGRAPH * 4);
  bf16*  gmA_b  = (bf16*)alloc((size_t)NGRAPH * 64 * 2);
  bf16*  gmB_b  = (bf16*)alloc((size_t)NGRAPH * 64 * 2);
  float* hh_f   = (float*)alloc((size_t)NGRAPH * 64 * 4);
  int*   srcA   = (int*)alloc((size_t)NEDGES * 4);
  int*   dstA   = (int*)alloc((size_t)NEDGES * 4);
  int*   egidx  = (int*)alloc((size_t)NEDGES * 4);

  auto gemm = [&](int nseg, Seg s0, Seg s1, Seg s2, Seg s3, const bf16* Wp,
                  const float* bias, float* outF, bf16* outB, int M) {
    GemmP p{};
    p.seg[0] = s0; p.seg[1] = s1; p.seg[2] = s2; p.seg[3] = s3;
    p.Wp = Wp; p.bias = bias; p.outF = outF; p.outB = outB; p.M = M;
    int blocks = ((M + 15) / 16 + 3) / 4;
    int om = (outF ? 1 : 0) | (outB ? 2 : 0);
    switch (nseg * 4 + om) {
      case 1 * 4 + 1: k_gemm<1, 1><<<blocks, 128, 0, stream>>>(p); break;
      case 1 * 4 + 2: k_gemm<1, 2><<<blocks, 128, 0, stream>>>(p); break;
      case 1 * 4 + 3: k_gemm<1, 3><<<blocks, 128, 0, stream>>>(p); break;
      case 3 * 4 + 2: k_gemm<3, 2><<<blocks, 128, 0, stream>>>(p); break;
      case 4 * 4 + 2: k_gemm<4, 2><<<blocks, 128, 0, stream>>>(p); break;
      default:        k_gemm<4, 3><<<blocks, 128, 0, stream>>>(p); break;
    }
  };
  Seg Z{nullptr, nullptr};

  // input conversion (zero-padded to 64 cols) + edge index split
  k_f2b_pad<<<cdiv((long)NNODES * 64, 256), 256, 0, stream>>>(node_features, nfin_b, NNODES, 64);
  k_f2b_pad<<<cdiv((long)NEDGES * 64, 256), 256, 0, stream>>>(edge_features, efin_b, NEDGES, 32);
  k_f2b_pad<<<cdiv((long)NGRAPH * 64, 256), 256, 0, stream>>>(global_features, gfin_b, NGRAPH, 16);
  k_split_edges<<<cdiv(NEDGES, 256), 256, 0, stream>>>(edge_indices, graph_indices, srcA, dstA, egidx, NEDGES);

  // embedding MLPs
  gemm(1, Seg{nfin_b, nullptr}, Z, Z, Z, pw_en1, F(6),  nullptr, hid_b, NNODES);
  gemm(1, Seg{hid_b,  nullptr}, Z, Z, Z, pw_en2, F(8),  nf_f,    nf_b,  NNODES);
  gemm(1, Seg{efin_b, nullptr}, Z, Z, Z, pw_ee1, F(10), nullptr, hid_b, NEDGES);
  gemm(1, Seg{hid_b,  nullptr}, Z, Z, Z, pw_ee2, F(12), ef_f,    ef_b,  NEDGES);
  gemm(1, Seg{gfin_b, nullptr}, Z, Z, Z, pw_eg1, F(14), nullptr, hid_b, NGRAPH);
  gemm(1, Seg{hid_b,  nullptr}, Z, Z, Z, pw_eg2, F(16), gf_f,    gf_b,  NGRAPH);

  for (int i = 0; i < 3; i++) {
    int pb = 17 + 12 * i;
    // edge MLP: e_in = [nf[src], nf[dst], ef, gf[egidx]]  (gather fused into GEMM)
    gemm(4, Seg{nf_b, srcA}, Seg{nf_b, dstA}, Seg{ef_b, nullptr}, Seg{gf_b, egidx},
         pw_be1[i], F(pb + 1), nullptr, hid_b, NEDGES);
    gemm(1, Seg{hid_b, nullptr}, Z, Z, Z, pw_be2[i], F(pb + 3), e_new, nullptr, NEDGES);
    // msgs = seg_mean(e_new, src)
    k_zero<<<cdiv((long)NNODES * 64, 256), 256, 0, stream>>>(nsum, (long)NNODES * 64);
    k_zero<<<cdiv(NNODES, 256), 256, 0, stream>>>(ncnt, NNODES);
    k_scat<<<cdiv((long)NEDGES * 64, 256), 256, 0, stream>>>(e_new, srcA, nsum, ncnt, NEDGES);
    k_norm<<<cdiv((long)NNODES * 64, 256), 256, 0, stream>>>(nsum, ncnt, nullptr, msgs_b, NNODES);
    // node MLP: n_in = [nf, msgs, gf[graph_indices]]
    gemm(3, Seg{nf_b, nullptr}, Seg{msgs_b, nullptr}, Seg{gf_b, graph_indices}, Z,
         pw_bn1[i], F(pb + 5), nullptr, hid_b, NNODES);
    gemm(1, Seg{hid_b, nullptr}, Z, Z, Z, pw_bn2[i], F(pb + 7), n_new, nullptr, NNODES);
    // graph-level means of n_new and e_new
    k_zero<<<cdiv((long)NGRAPH * 64, 256), 256, 0, stream>>>(gsumA, (long)NGRAPH * 64);
    k_zero<<<1, 256, 0, stream>>>(gcntA, NGRAPH);
    k_zero<<<cdiv((long)NGRAPH * 64, 256), 256, 0, stream>>>(gsumB, (long)NGRAPH * 64);
    k_zero<<<1, 256, 0, stream>>>(gcntB, NGRAPH);
    k_scat<<<cdiv((long)NNODES * 64, 256), 256, 0, stream>>>(n_new, graph_indices, gsumA, gcntA, NNODES);
    k_scat<<<cdiv((long)NEDGES * 64, 256), 256, 0, stream>>>(e_new, egidx, gsumB, gcntB, NEDGES);
    k_norm<<<cdiv((long)NGRAPH * 64, 256), 256, 0, stream>>>(gsumA, gcntA, nullptr, gmA_b, NGRAPH);
    k_norm<<<cdiv((long)NGRAPH * 64, 256), 256, 0, stream>>>(gsumB, gcntB, nullptr, gmB_b, NGRAPH);
    // glob MLP: g_in = [node_msg, edge_msg, gf]
    gemm(3, Seg{gmA_b, nullptr}, Seg{gmB_b, nullptr}, Seg{gf_b, nullptr}, Z,
         pw_bg1[i], F(pb + 9), nullptr, hid_b, NGRAPH);
    gemm(1, Seg{hid_b, nullptr}, Z, Z, Z, pw_bg2[i], F(pb + 11), g_new, nullptr, NGRAPH);
    // residual updates (f32 state + bf16 mirrors)
    k_resid<<<cdiv((long)NNODES * 64, 256), 256, 0, stream>>>(nf_f, n_new, nf_b, (long)NNODES * 64);
    k_resid<<<cdiv((long)NEDGES * 64, 256), 256, 0, stream>>>(ef_f, e_new, ef_b, (long)NEDGES * 64);
    k_resid<<<cdiv((long)NGRAPH * 64, 256), 256, 0, stream>>>(gf_f, g_new, gf_b, (long)NGRAPH * 64);
  }

  // readout: graph_feat = [seg_mean(nf), seg_mean(ef), gf]
  k_zero<<<cdiv((long)NGRAPH * 64, 256), 256, 0, stream>>>(gsumA, (long)NGRAPH * 64);
  k_zero<<<1, 256, 0, stream>>>(gcntA, NGRAPH);
  k_zero<<<cdiv((long)NGRAPH * 64, 256), 256, 0, stream>>>(gsumB, (long)NGRAPH * 64);
  k_zero<<<1, 256, 0, stream>>>(gcntB, NGRAPH);
  k_scat<<<cdiv((long)NNODES * 64, 256), 256, 0, stream>>>(nf_f, graph_indices, gsumA, gcntA, NNODES);
  k_scat<<<cdiv((long)NEDGES * 64, 256), 256, 0, stream>>>(ef_f, egidx, gsumB, gcntB, NEDGES);
  k_norm<<<cdiv((long)NGRAPH * 64, 256), 256, 0, stream>>>(gsumA, gcntA, nullptr, gmA_b, NGRAPH);
  k_norm<<<cdiv((long)NGRAPH * 64, 256), 256, 0, stream>>>(gsumB, gcntB, nullptr, gmB_b, NGRAPH);
  gemm(3, Seg{gmA_b, nullptr}, Seg{gmB_b, nullptr}, Seg{gf_b, nullptr}, Z,
       pw_o1, F(54), nullptr, hid_b, NGRAPH);
  gemm(1, Seg{hid_b, nullptr}, Z, Z, Z, pw_o2, F(56), hh_f, nullptr, NGRAPH);
  k_head3<<<1, 256, 0, stream>>>(hh_f, F(57), F(58), (float*)d_out, NGRAPH);
}
